// AttentionWithRotaryPositionalEmbedding_50706383896938
// MI455X (gfx1250) — matrix operand order repaired and checked by reading, not compiled, not run on hardware
//
#include <hip/hip_runtime.h>

// ---------------- problem constants ----------------
#define BSZ  4
#define SEQ  2048
#define CH   512
#define NH   8
#define DH   64
#define QT   64     // q rows per workgroup (4 waves x 16)
#define KT   64     // keys per K/V tile
#define LDK  72     // padded LDS row stride (halves): 64 + 8 (TDM pad 4 dwords)

typedef __attribute__((ext_vector_type(16))) _Float16 v16h;
typedef __attribute__((ext_vector_type(8)))  _Float16 v8h;
typedef __attribute__((ext_vector_type(8)))  float    v8f;
typedef __attribute__((ext_vector_type(4)))  unsigned v4u;
typedef __attribute__((ext_vector_type(8)))  int      v8i;
typedef __attribute__((ext_vector_type(4)))  int      v4i;

// ---------------- WMMA helpers ----------------
__device__ __forceinline__ v8f wmma_f16(v16h a, v16h b, v8f c) {
  return __builtin_amdgcn_wmma_f32_16x16x32_f16(false, a, false, b, (short)0, c,
                                                false, false);
}

// A fragment 16x32 (MxK): lane<16 holds row (lane&15), K = k0+{0..7, 16..23};
// lane>=16 same row, K = k0+{8..15, 24..31}.  Two 16B loads.
__device__ __forceinline__ v16h ldfragA(const _Float16* base, int stride,
                                        int lane, int k0) {
  const _Float16* p = base + (lane & 15) * stride + k0 + ((lane & 16) ? 8 : 0);
  v8h lo = *(const v8h*)(p);
  v8h hi = *(const v8h*)(p + 16);
  return __builtin_shufflevector(lo, hi, 0,1,2,3,4,5,6,7,8,9,10,11,12,13,14,15);
}

// B fragment 32x16 (KxN), sourced from a Bt[n][k] (row = output col) layout:
// lane<16 holds col (lane&15), K = k0+{0..7, 16..23}; lane>=16: K = k0+{8..15, 24..31}.
__device__ __forceinline__ v16h ldfragB(const _Float16* base, int stride,
                                        int lane, int k0) {
  const _Float16* p = base + (lane & 15) * stride + k0 + ((lane & 16) ? 8 : 0);
  v8h lo = *(const v8h*)(p);
  v8h hi = *(const v8h*)(p + 16);
  return __builtin_shufflevector(lo, hi, 0,1,2,3,4,5,6,7,8,9,10,11,12,13,14,15);
}

// ---------------- TDM 2-D tile load (global -> LDS) ----------------
// Loads tile (tl1 rows x tl0 elems) of 2-byte data, row stride s0 (elems),
// into LDS at lds_off with 4-dword padding per 32-dword row (-> 72-half rows).
__device__ __forceinline__ void tdm_load_2d(unsigned lds_off, const _Float16* g,
                                            unsigned td0, unsigned td1,
                                            unsigned tl0, unsigned tl1,
                                            unsigned long long s0) {
  unsigned long long ga = (unsigned long long)(size_t)g;
  v4u g0;
  g0[0] = 1u;                                        // count=1, user mode
  g0[1] = lds_off;                                   // LDS byte address
  g0[2] = (unsigned)(ga & 0xFFFFFFFFu);              // global_addr[31:0]
  g0[3] = (unsigned)((ga >> 32) & 0x01FFFFFFu) | (2u << 30);  // addr hi | type=2
  v8i g1;
  g1[0] = (int)((1u << 16)      // data_size = 2 bytes
              | (1u << 20)      // pad_enable
              | (4u << 22)      // pad_interval: 32 dwords (one 64-half row)
              | (3u << 25));    // pad_amount: 4 dwords (8 halves)
  g1[1] = (int)((td0 & 0xFFFFu) << 16);              // tensor_dim0 lo
  g1[2] = (int)((td0 >> 16) | ((td1 & 0xFFFFu) << 16));
  g1[3] = (int)((td1 >> 16) | (tl0 << 16));          // tile_dim0
  g1[4] = (int)(tl1 & 0xFFFFu);                      // tile_dim1 (tile_dim2=0)
  g1[5] = (int)(s0 & 0xFFFFFFFFull);                 // tensor_dim0_stride lo
  g1[6] = (int)((s0 >> 32) & 0xFFFFull);             // stride hi (dim1_stride=0)
  g1[7] = 0;
  v4i z4 = {0, 0, 0, 0};
  v8i z8 = {0, 0, 0, 0, 0, 0, 0, 0};
  // clang-23 / therock-10.0 signature: (v4u, v8i, v4i, v4i, v8i, i32 cpol)
  __builtin_amdgcn_tensor_load_to_lds(g0, g1, z4, z4, z8, 0);
}

// ---------------- kernel 1: fp32 -> f16 converts ----------------
__global__ void cvt_kernel(const float* __restrict__ x,
                           const float* __restrict__ wqkv,
                           const float* __restrict__ wproj,
                           _Float16* __restrict__ xh,
                           _Float16* __restrict__ wqkvh,
                           _Float16* __restrict__ wprojh) {
  const int NX = BSZ * SEQ * CH;        // 4194304
  const int NW1 = 3 * CH * CH;          // 786432
  const int NW2 = CH * CH;              // 262144
  int i = blockIdx.x * 256 + threadIdx.x;
  if (i < NX)                xh[i] = (_Float16)x[i];
  else if (i < NX + NW1)     wqkvh[i - NX] = (_Float16)wqkv[i - NX];
  else if (i < NX + NW1 + NW2) wprojh[i - NX - NW1] = (_Float16)wproj[i - NX - NW1];
}

// ---------------- kernel 2: QKV GEMM + RoPE + per-head scatter ----------------
// out(m,n) = sum_k xh(m,k) * Wqkv(n,k);  M=8192, N=1536, K=512
// wave computes 16x64; WG (128 thr) computes 64x64.
__global__ __launch_bounds__(128) void qkv_rope_kernel(
    const _Float16* __restrict__ xh, const _Float16* __restrict__ wqkv,
    const float* __restrict__ times, const int* __restrict__ nclsp,
    _Float16* __restrict__ qh, _Float16* __restrict__ kh,
    _Float16* __restrict__ vT) {
  int wave = threadIdx.x >> 5, lane = threadIdx.x & 31;
  int mbase = blockIdx.x * 64 + wave * 16;
  int n0 = blockIdx.y * 64;
  const _Float16* A  = xh + (size_t)mbase * CH;
  const _Float16* Bt = wqkv + (size_t)n0 * CH;

  v8f acc[4] = {{}, {}, {}, {}};
  for (int k = 0; k < CH; k += 32) {
    v16h a = ldfragA(A, CH, lane, k);
#pragma unroll
    for (int t = 0; t < 4; ++t)
      acc[t] = wmma_f16(a, ldfragB(Bt + t * 16 * CH, CH, lane, k), acc[t]);
  }

  int ncls = *nclsp;
  int cl = lane & 15, ro = (lane & 16) ? 8 : 0;
#pragma unroll
  for (int t = 0; t < 4; ++t) {
    int n = n0 + t * 16 + cl;
    int which = n / CH;                 // 0=q 1=k 2=v (uniform per subtile)
    int c = n - which * CH;
    int h = c / DH, d = c & (DH - 1);
    // inv_freq = 10000^(-2*(d/2)/64)
    float invf = __expf(-9.210340371976184f * (float)(2 * (d >> 1)) * (1.0f / 64.0f));
#pragma unroll
    for (int e = 0; e < 8; ++e) {
      int m = mbase + ro + e;
      int bb = m >> 11, tok = m & (SEQ - 1);
      float val = acc[t][e];
      float partner = __shfl_xor(val, 1);   // paired d column lives in lane^1
      if (which == 2) {
        vT[(((size_t)bb * NH + h) * DH + d) * SEQ + tok] = (_Float16)val;
      } else {
        if (tok >= ncls) {
          float pos = rintf(times[bb * SEQ + (tok - ncls)] * 30.0f);
          float ang = pos * invf;
          float s_, c_;
          __sincosf(ang, &s_, &c_);
          val = (d & 1) ? fmaf(partner, s_, val * c_)
                        : fmaf(-partner, s_, val * c_);
        }
        _Float16* dst = (which == 0) ? qh : kh;
        dst[(((size_t)bb * NH + h) * SEQ + tok) * DH + d] = (_Float16)val;
      }
    }
  }
}

// ---------------- kernel 3: flash attention with TDM-fed LDS tiles ----------
__global__ __launch_bounds__(128) void attn_kernel(
    const _Float16* __restrict__ qh, const _Float16* __restrict__ kh,
    const _Float16* __restrict__ vT, const float* __restrict__ mask,
    _Float16* __restrict__ oh) {
  __shared__ __align__(16) _Float16 skv[2][2][KT * LDK];  // [buf][K|V] 36 KB
  __shared__ __align__(16) _Float16 sp[4][16 * LDK];      // P restage, 9 KB

  int bh = blockIdx.x, qb = blockIdx.y;
  int b = bh >> 3, h = bh & 7;
  int wave = threadIdx.x >> 5, lane = threadIdx.x & 31;

  const _Float16* qbase = qh + ((size_t)bh * SEQ + qb * QT + wave * 16) * DH;
  const _Float16* kbase = kh + (size_t)bh * SEQ * DH;
  const _Float16* vbase = vT + (size_t)bh * DH * SEQ;
  const float* mrow = mask + b * SEQ;

  v16h aq0 = ldfragA(qbase, DH, lane, 0);
  v16h aq1 = ldfragA(qbase, DH, lane, 32);

  v8f o[4] = {{}, {}, {}, {}};
  float m_e[8], l_e[8];
#pragma unroll
  for (int e = 0; e < 8; ++e) { m_e[e] = -3.0e38f; l_e[e] = 0.0f; }

  const int NT = SEQ / KT;  // 32 tiles
  if (wave == 0) {
    tdm_load_2d((unsigned)(size_t)&skv[0][0][0], kbase, DH, SEQ, DH, KT, DH);
    tdm_load_2d((unsigned)(size_t)&skv[0][1][0], vbase, SEQ, DH, KT, DH, SEQ);
  }
  for (int jt = 0; jt < NT; ++jt) {
    if (wave == 0) {
      if (jt + 1 < NT) {  // prefetch next tile into the other buffer
        int nb = (jt + 1) & 1;
        tdm_load_2d((unsigned)(size_t)&skv[nb][0][0],
                    kbase + (size_t)(jt + 1) * KT * DH, DH, SEQ, DH, KT, DH);
        tdm_load_2d((unsigned)(size_t)&skv[nb][1][0],
                    vbase + (jt + 1) * KT, SEQ, DH, KT, DH, SEQ);
        __builtin_amdgcn_s_wait_tensorcnt((short)2);  // current tile done
      } else {
        __builtin_amdgcn_s_wait_tensorcnt((short)0);
      }
    }
    __syncthreads();
    const _Float16* Ktile = &skv[jt & 1][0][0];
    const _Float16* Vtile = &skv[jt & 1][1][0];

    // S = Q K^T  (16 rows x 64 keys per wave)
    v8f s[4] = {{}, {}, {}, {}};
#pragma unroll
    for (int t = 0; t < 4; ++t) {
      s[t] = wmma_f16(aq0, ldfragB(Ktile + t * 16 * LDK, LDK, lane, 0), s[t]);
      s[t] = wmma_f16(aq1, ldfragB(Ktile + t * 16 * LDK, LDK, lane, 32), s[t]);
    }
    // scale + additive mask
#pragma unroll
    for (int t = 0; t < 4; ++t) {
      float mv = mrow[jt * KT + t * 16 + (lane & 15)];
#pragma unroll
      for (int e = 0; e < 8; ++e) s[t][e] = s[t][e] * 0.125f + mv;
    }
    // online softmax: rows live in elements; 16-lane half-wave reductions
#pragma unroll
    for (int e = 0; e < 8; ++e) {
      float mx = fmaxf(fmaxf(s[0][e], s[1][e]), fmaxf(s[2][e], s[3][e]));
      mx = fmaxf(mx, __shfl_xor(mx, 1));
      mx = fmaxf(mx, __shfl_xor(mx, 2));
      mx = fmaxf(mx, __shfl_xor(mx, 4));
      mx = fmaxf(mx, __shfl_xor(mx, 8));
      float mn = fmaxf(m_e[e], mx);
      float alpha = __expf(m_e[e] - mn);
      m_e[e] = mn;
      float rs = 0.0f;
#pragma unroll
      for (int t = 0; t < 4; ++t) { s[t][e] = __expf(s[t][e] - mn); rs += s[t][e]; }
      rs += __shfl_xor(rs, 1);
      rs += __shfl_xor(rs, 2);
      rs += __shfl_xor(rs, 4);
      rs += __shfl_xor(rs, 8);
      l_e[e] = l_e[e] * alpha + rs;
#pragma unroll
      for (int t = 0; t < 4; ++t) o[t][e] *= alpha;
    }
    // restage P (D-layout) into LDS as f16 A-layout source (per-wave region;
    // DS ops are in-order within a wave, no barrier needed)
    _Float16* prow = &sp[wave][0];
    {
      int cl = lane & 15, ro = (lane & 16) ? 8 : 0;
#pragma unroll
      for (int t = 0; t < 4; ++t)
#pragma unroll
        for (int e = 0; e < 8; ++e)
          prow[(ro + e) * LDK + t * 16 + cl] = (_Float16)s[t][e];
    }
    v16h p0 = ldfragA(prow, LDK, lane, 0);
    v16h p1 = ldfragA(prow, LDK, lane, 32);
    // O += P V   (V tile is [d][key] so rows are output columns)
#pragma unroll
    for (int t = 0; t < 4; ++t) {
      o[t] = wmma_f16(p0, ldfragB(Vtile + t * 16 * LDK, LDK, lane, 0), o[t]);
      o[t] = wmma_f16(p1, ldfragB(Vtile + t * 16 * LDK, LDK, lane, 32), o[t]);
    }
    __syncthreads();  // done reading buffer before TDM overwrites it
  }
  // epilogue: normalize, write (b, n, h*DH + d) as f16 for proj GEMM
  int cl = lane & 15, ro = (lane & 16) ? 8 : 0;
#pragma unroll
  for (int e = 0; e < 8; ++e) {
    float inv = 1.0f / l_e[e];
    int n = qb * QT + wave * 16 + ro + e;
    _Float16* dst = oh + ((size_t)b * SEQ + n) * CH + h * DH;
#pragma unroll
    for (int t = 0; t < 4; ++t) dst[t * 16 + cl] = (_Float16)(o[t][e] * inv);
  }
}

// ---------------- kernel 4: output projection + bias ----------------
__global__ __launch_bounds__(128) void proj_kernel(
    const _Float16* __restrict__ oh, const _Float16* __restrict__ wp,
    const float* __restrict__ bias, float* __restrict__ out) {
  int wave = threadIdx.x >> 5, lane = threadIdx.x & 31;
  int mbase = blockIdx.x * 64 + wave * 16;
  int n0 = blockIdx.y * 64;
  const _Float16* A  = oh + (size_t)mbase * CH;
  const _Float16* Bt = wp + (size_t)n0 * CH;
  v8f acc[4] = {{}, {}, {}, {}};
  for (int k = 0; k < CH; k += 32) {
    v16h a = ldfragA(A, CH, lane, k);
#pragma unroll
    for (int t = 0; t < 4; ++t)
      acc[t] = wmma_f16(a, ldfragB(Bt + t * 16 * CH, CH, lane, k), acc[t]);
  }
  int cl = lane & 15, ro = (lane & 16) ? 8 : 0;
#pragma unroll
  for (int t = 0; t < 4; ++t) {
    int n = n0 + t * 16 + cl;
    float bv = bias[n];
#pragma unroll
    for (int e = 0; e < 8; ++e)
      out[(size_t)(mbase + ro + e) * CH + n] = acc[t][e] + bv;
  }
}

// ---------------- launch ----------------
extern "C" void kernel_launch(void* const* d_in, const int* in_sizes, int n_in,
                              void* d_out, int out_size, void* d_ws,
                              size_t ws_size, hipStream_t stream) {
  const float* x     = (const float*)d_in[0];
  const float* mask  = (const float*)d_in[1];
  const float* times = (const float*)d_in[2];
  const float* Wqkv  = (const float*)d_in[3];
  const float* Wproj = (const float*)d_in[4];
  const float* bproj = (const float*)d_in[5];
  const int*   ncls  = (const int*)d_in[6];

  const size_t NX = (size_t)BSZ * SEQ * CH;  // 4194304 elems
  _Float16* xh     = (_Float16*)d_ws;
  _Float16* wqkvh  = xh + NX;
  _Float16* wprojh = wqkvh + (size_t)3 * CH * CH;
  _Float16* qh     = wprojh + (size_t)CH * CH;
  _Float16* kh     = qh + NX;
  _Float16* vT     = kh + NX;
  _Float16* oh     = vT + NX;

  const int ncvt = (int)(NX + 3 * CH * CH + CH * CH);
  cvt_kernel<<<(ncvt + 255) / 256, 256, 0, stream>>>(x, Wqkv, Wproj, xh, wqkvh,
                                                     wprojh);
  qkv_rope_kernel<<<dim3((BSZ * SEQ) / 64, (3 * CH) / 64), 128, 0, stream>>>(
      xh, wqkvh, times, ncls, qh, kh, vT);
  attn_kernel<<<dim3(BSZ * NH, SEQ / QT), 128, 0, stream>>>(qh, kh, vT, mask,
                                                            oh);
  proj_kernel<<<dim3((BSZ * SEQ) / 64, CH / 64), 128, 0, stream>>>(
      oh, wprojh, bproj, (float*)d_out);
}
